// RadiallyParamaterisedTensorProduct_8203387535828
// MI455X (gfx1250) — compile-verified, matrix operand order checked by reading, not run on hardware
//
#include <hip/hip_runtime.h>
#include <cstdint>
#include <cstddef>

// ---------------------------------------------------------------------------
// MI455X (gfx1250) fused RadiallyParameterisedTensorProduct
//   out(B,288) = TP( feats(B,128), geo(B,9), weights ),
//   weights(B,7168) = silu(norm*w1+b1)(B,64) @ w2^T           <-- 27.5 GFLOP
// Fully fused: the 860MB `weights` tensor never leaves registers.
// GEMM runs on v_wmma_f32_16x16x32_f16 (wave32, 16 samples / wave tile).
// Round 3: amdgpu_waves_per_eu(4) (real VGPR cap: 1024/4=256) + explicit
// unroll_count(2) on the K-outer loop so the pipeliner doesn't balloon
// liveness to ~900 VGPRs / 1 wave/SIMD like the full-unroll did.
// ---------------------------------------------------------------------------

typedef __attribute__((ext_vector_type(16))) _Float16 v16h;
typedef __attribute__((ext_vector_type(8)))  float    v8f;

__device__ __forceinline__ int imin(int a, int b) { return a < b ? a : b; }

// K-offset of half element i (0..15) in a 16-bit WMMA operand register set,
// per CDNA5 ISA 7.12.2 "16-bit A-Matrix 16x32" (B assumed mirrored, lane->N).
__device__ __forceinline__ int frag_k(int lane, int i) {
    int v = i >> 1, h = i & 1;
    return ((v < 4) ? 0 : 16) + ((lane < 16) ? 0 : 8) + ((v & 3) << 1) + h;
}

// ---------------------------------------------------------------------------
// Pre-kernel: convert w2 (7168x64 f32) into f16 B-fragments in d_ws.
// Fragment id f = (p*64 + t)*2 + ks;  t = 2*u + whi covers cols u*32+whi*16+n.
// Fragment storage: 512 halves, [lane(32)][i(16)] contiguous (2x b128 / lane).
// ---------------------------------------------------------------------------
#define NFRAG_ELEMS (7 * 64 * 2 * 512)   // == 458752 == 7168*64

__global__ __launch_bounds__(256) void w2_to_frags(const float* __restrict__ w2,
                                                   _Float16* __restrict__ wsh) {
    int idx = blockIdx.x * 256 + threadIdx.x;
    if (idx >= NFRAG_ELEMS) return;
    int i    = idx & 15;
    int lane = (idx >> 4) & 31;
    int f    = idx >> 9;
    int ks   = f & 1;
    int t    = (f >> 1) & 63;
    int p    = f >> 7;
    int n    = lane & 15;
    int k    = frag_k(lane, i);
    int u    = t >> 1, whi = t & 1;
    int col  = u * 32 + whi * 16 + n;           // GEMM column = u*32 + w
    int row  = p * 1024 + col;                  // row of w2
    wsh[idx] = (_Float16)w2[row * 64 + ks * 32 + k];
}

// ---------------------------------------------------------------------------
// Main fused kernel: 1 wave (32 lanes) per 16-sample tile.
// amdgpu_waves_per_eu(4): cap 1024/4 = 256 VGPRs => direct VGPR addressing,
// 4 waves resident per SIMD32 for latency hiding.
// ---------------------------------------------------------------------------
__global__ __launch_bounds__(32)
__attribute__((amdgpu_waves_per_eu(4)))
void rptp_fused(
    const float* __restrict__ feats, const float* __restrict__ geo,
    const float* __restrict__ norm,  const float* __restrict__ w1,
    const float* __restrict__ b1,    const float* __restrict__ b2,
    const _Float16* __restrict__ wfrag, float* __restrict__ out, int B)
{
    __shared__ float tmp_lds[16 * 32 * 25];     // 50 KB: tmp[m][u][21] pad->25

    const int lane  = threadIdx.x;              // wave32
    const int wlo   = lane & 15;                // lane's N column (w low bits)
    const int mbase = (lane >> 4) << 3;         // D rows: M = r + mbase
    const int z0    = blockIdx.x * 16;

    // ---- A fragments: h[m][j] = silu(norm[z]*w1[j]+b1[j]), m = lane&15 ----
    const float nm = norm[imin(z0 + wlo, B - 1)];
    union AF { v16h v; _Float16 e[16]; } a0, a1;
#pragma unroll
    for (int i = 0; i < 16; ++i) {
        int k = frag_k(lane, i);
        {   int j = k;                       // K-step 0: j = 0..31
            float x = nm * w1[j] + b1[j];
            a0.e[i] = (_Float16)(x / (1.0f + __expf(-x))); }
        {   int j = 32 + k;                  // K-step 1: j = 32..63
            float x = nm * w1[j] + b1[j];
            a1.e[i] = (_Float16)(x / (1.0f + __expf(-x))); }
    }

    // ---- tmp[m][u][*]: CG-contracted geometry, path-norm folded in ----
    // path norms: o=0: sqrt(1/64)  o=1: sqrt(3/96)  o=2: sqrt(5/64)
    const float PW0 = 0.125f, PW1 = 0.17677669529663687f, PW2 = 0.2795084971874737f;
    const float IS3 = 0.5773502691896258f, IS5 = 0.4472135954999579f;
    const float C10 = 0.31622776601683794f;   // 1/sqrt(10): w3j(1,1,2) entries
    {
        const int u = lane;                   // one channel per lane
        for (int m = 0; m < 16; ++m) {
            int zc = imin(z0 + m, B - 1);
            const float* g = geo + (size_t)zc * 9;
            float g0 = g[0];
            float v0 = g[1], v1 = g[2], v2 = g[3];                 // l=1 (y,z,x)
            float q0 = g[4], q1 = g[5], q2 = g[6], q3 = g[7], q4 = g[8]; // l=2
            const float* fz = feats + (size_t)zc * 128;
            float s  = fz[u];                                      // 32x0e
            float a_0 = fz[32 + u * 3 + 0];                        // 32x1o
            float a_1 = fz[32 + u * 3 + 1];
            float a_2 = fz[32 + u * 3 + 2];
            float* T = &tmp_lds[(m * 32 + u) * 25];
            T[0]  = PW0 * s * g0;                       // (0,0,0)
            T[1]  = PW1 * IS3 * s * v0;                 // (0,1,1)
            T[2]  = PW1 * IS3 * s * v1;
            T[3]  = PW1 * IS3 * s * v2;
            T[4]  = PW2 * IS5 * s * q0;                 // (0,2,2)
            T[5]  = PW2 * IS5 * s * q1;
            T[6]  = PW2 * IS5 * s * q2;
            T[7]  = PW2 * IS5 * s * q3;
            T[8]  = PW2 * IS5 * s * q4;
            T[9]  = PW1 * IS3 * a_0 * g0;               // (1,0,1)
            T[10] = PW1 * IS3 * a_1 * g0;
            T[11] = PW1 * IS3 * a_2 * g0;
            T[12] = PW0 * IS3 * (a_0 * v0 + a_1 * v1 + a_2 * v2);  // (1,1,0)
            // (1,1,2): real w3j entries +-1/sqrt(10)
            T[13] = PW2 * C10 * (a_2 * v0 + a_0 * v2);
            T[14] = PW2 * C10 * (a_0 * v1 + a_1 * v0);
            T[15] = PW2 * C10 * IS3 * (2.0f * a_1 * v1 - a_2 * v2 - a_0 * v0);
            T[16] = PW2 * C10 * (a_2 * v1 + a_1 * v2);
            T[17] = PW2 * C10 * (a_2 * v2 - a_0 * v0);
            // (1,2,1): same tensor, l=2 leg contracted with geo l=2
            T[18] = PW1 * C10 * (a_2 * q0 + a_1 * q1 - IS3 * a_0 * q2 - a_0 * q4);
            T[19] = PW1 * C10 * (a_0 * q1 + 2.0f * IS3 * a_1 * q2 + a_2 * q3);
            T[20] = PW1 * C10 * (a_0 * q0 - IS3 * a_2 * q2 + a_1 * q3 + a_2 * q4);
        }
    }
    __syncthreads();

    const int DKS[7]  = {1, 3, 5, 3, 1, 5, 3};   // k-width per path
    const int KOFF[7] = {0, 1, 4, 1, 0, 4, 1};   // acc slot (o0:0, o1:1-3, o2:4-8)
    const int TCOL[7] = {0, 1, 4, 9, 12, 13, 18};

    // whi outermost: halves live accumulators (acc[8][9] = 72 VGPRs)
    for (int whi = 0; whi < 2; ++whi) {
        float acc[8][9];
#pragma unroll
        for (int r = 0; r < 8; ++r)
#pragma unroll
            for (int k = 0; k < 9; ++k) acc[r][k] = 0.0f;

#pragma unroll
        for (int p = 0; p < 7; ++p) {
            const int dk = DKS[p], ko = KOFF[p], tc = TCOL[p];
            const _Float16* pfrag = wfrag + (size_t)p * 64 * 2 * 512;
            const float*    pb2   = b2 + p * 1024;
#pragma clang loop unroll_count(2)
            for (int u = 0; u < 32; ++u) {
                // broadcast tmp for this wave-half's 8 samples
                float tm[8][5];
#pragma unroll
                for (int r = 0; r < 8; ++r)
#pragma unroll
                    for (int kk = 0; kk < 5; ++kk)
                        if (kk < dk)
                            tm[r][kk] = tmp_lds[((mbase + r) * 32 + u) * 25 + tc + kk];

                const int t = u * 2 + whi;
                union BF { v16h v; uint4 q[2]; } bf0, bf1;
                const uint4* s0 = (const uint4*)(pfrag + ((size_t)(t * 2 + 0) * 512 + lane * 16));
                bf0.q[0] = s0[0]; bf0.q[1] = s0[1];
                const uint4* s1 = (const uint4*)(pfrag + ((size_t)(t * 2 + 1) * 512 + lane * 16));
                bf1.q[0] = s1[0]; bf1.q[1] = s1[1];

                v8f d = {0.f, 0.f, 0.f, 0.f, 0.f, 0.f, 0.f, 0.f};
                d = __builtin_amdgcn_wmma_f32_16x16x32_f16(false, a0.v, false, bf0.v,
                                                           (short)0, d, false, false);
                d = __builtin_amdgcn_wmma_f32_16x16x32_f16(false, a1.v, false, bf1.v,
                                                           (short)0, d, false, false);
                // lane's weight column = u*32 + whi*16 + wlo; add bias, contract u
                float b2v = pb2[u * 32 + whi * 16 + wlo];
#pragma unroll
                for (int r = 0; r < 8; ++r) {
                    float wv = d[r] + b2v;
#pragma unroll
                    for (int kk = 0; kk < 5; ++kk)
                        if (kk < dk) acc[r][ko + kk] += wv * tm[r][kk];
                }
            }
        }

        // ---- store: out row = [32x0e | 32x1o (w*3+k) | 32x2e (w*5+k)] ----
        const int w = whi * 16 + wlo;
#pragma unroll
        for (int r = 0; r < 8; ++r) {
            int z = z0 + mbase + r;
            if (z < B) {
                float* o = out + (size_t)z * 288;
                o[w]                 = acc[r][0];
                o[32  + w * 3 + 0]   = acc[r][1];
                o[32  + w * 3 + 1]   = acc[r][2];
                o[32  + w * 3 + 2]   = acc[r][3];
                o[128 + w * 5 + 0]   = acc[r][4];
                o[128 + w * 5 + 1]   = acc[r][5];
                o[128 + w * 5 + 2]   = acc[r][6];
                o[128 + w * 5 + 3]   = acc[r][7];
                o[128 + w * 5 + 4]   = acc[r][8];
            }
        }
    }
}

// ---------------------------------------------------------------------------
extern "C" void kernel_launch(void* const* d_in, const int* in_sizes, int n_in,
                              void* d_out, int out_size, void* d_ws, size_t ws_size,
                              hipStream_t stream) {
    const float* feats = (const float*)d_in[0];   // (B,128)
    const float* geo   = (const float*)d_in[1];   // (B,9)
    const float* norm  = (const float*)d_in[2];   // (B,1)
    const float* w1    = (const float*)d_in[3];   // (64,1)
    const float* b1    = (const float*)d_in[4];   // (64,)
    const float* w2    = (const float*)d_in[5];   // (7168,64)
    const float* b2    = (const float*)d_in[6];   // (7168,)
    float* out = (float*)d_out;
    const int B = in_sizes[0] / 128;

    // Stage w2 as f16 WMMA B-fragments in workspace (917504 bytes needed).
    _Float16* wsh = (_Float16*)d_ws;
    w2_to_frags<<<(NFRAG_ELEMS + 255) / 256, 256, 0, stream>>>(w2, wsh);

    const int tiles = (B + 15) / 16;              // 1875 for B=30000
    rptp_fused<<<tiles, 32, 0, stream>>>(feats, geo, norm, w1, b1, b2, wsh, out, B);
}
